// HeteroGNN_17093969838497
// MI455X (gfx1250) — matrix operand order repaired and checked by reading, not compiled, not run on hardware
//
#include <hip/hip_runtime.h>
#include <hip/hip_bf16.h>

// ---------------------------------------------------------------------------
// CDNA5 (gfx1250) hetero-GraphSAGE pipeline.
//  - GEMMs: v_wmma_f32_16x16x32_bf16, fp32 accumulate, bias+ReLU/leaky fused.
//  - Intermediate activations (h, f) stored bf16 -> halves dominant
//    gather/GEMM-activation HBM traffic (roofline: scatter/gather bound).
//  - bf16 A-tiles staged with GLOBAL_LOAD_ASYNC_TO_LDS_B128 (inline asm,
//    ASYNCcnt) -- no conversion needed, pure 16B/lane DMA into LDS.
//  - Edge aggregation: bf16 gather + fp32 global_atomic_add_f32 scatter.
// ---------------------------------------------------------------------------

typedef __attribute__((ext_vector_type(16))) __bf16 v16bf;
typedef __attribute__((ext_vector_type(8)))  float  v8f;

union Frag16 {
    v16bf bf;
    uint4 q[2];
};

__device__ __forceinline__ unsigned short f2bf(float x) {
    union { float f; unsigned u; } v;
    v.f = x;
    unsigned r = v.u + 0x7fffu + ((v.u >> 16) & 1u);   // round-to-nearest-even
    return (unsigned short)(r >> 16);
}

__device__ __forceinline__ unsigned packbf2(float a, float b) {
    return (unsigned)f2bf(a) | ((unsigned)f2bf(b) << 16);
}

__device__ __forceinline__ float bflo(unsigned w) {   // low bf16 of packed word
    union { float f; unsigned u; } v; v.u = w << 16; return v.f;
}
__device__ __forceinline__ float bfhi(unsigned w) {   // high bf16 of packed word
    union { float f; unsigned u; } v; v.u = w & 0xffff0000u; return v.f;
}

// ---------------------------------------------------------------------------
// C_bf16[M,N] = act( A1f[M,K1] @ W1[K1,N] + A2bf[M,K2] @ W2[K2,N] + bias[N] )
// N == 256. slope: 0 => ReLU, 0.01 => leaky. A1 fp32, A2 bf16 (may be null).
// Block: 256 threads (8 waves). Block tile 64x64. Wave tile 16x32 (2 WMMAs).
// ---------------------------------------------------------------------------
__global__ void __launch_bounds__(256)
wmma_gemm_bias_act(const float* __restrict__ A1, const float* __restrict__ W1, int K1,
                   const unsigned short* __restrict__ A2, const float* __restrict__ W2, int K2,
                   const float* __restrict__ bias, unsigned short* __restrict__ C,
                   int M, int N, float slope)
{
    // 64 rows x 32 k (A) and 64 cols x 32 k (B, transposed). Row stride 40
    // shorts = 80 bytes: 16B-aligned vectors, conflict-free b128 reads.
    __shared__ __align__(16) unsigned short Alds[64 * 40];
    __shared__ __align__(16) unsigned short Blds[64 * 40];

    const int rowBase = blockIdx.y * 64;
    const int colBase = blockIdx.x * 64;
    const int t    = threadIdx.x;
    const int wave = t >> 5;
    const int lane = t & 31;
    const int waveR = wave & 3;      // 0..3  -> 16-row strip
    const int waveC = wave >> 2;     // 0..1  -> 32-col strip
    const int half = lane >> 4;      // 0/1
    const int l16  = lane & 15;

    v8f acc0 = {};
    v8f acc1 = {};

    for (int phase = 0; phase < 2; ++phase) {
        const float* W = phase ? W2 : W1;
        const int    K = phase ? K2 : K1;
        if (K == 0) continue;

        for (int k0 = 0; k0 < K; k0 += 32) {
            if (phase == 0) {
                // ---- stage fp32 A tile: 64x32 -> bf16, 8 elems/thread ----
                const int r  = t >> 2;            // 0..63
                const int c8 = (t & 3) << 3;      // 0,8,16,24
                const int grow = rowBase + r;
                float v0, v1, v2, v3, v4, v5, v6, v7;
                if (grow < M) {
                    const float* p = A1 + (long)grow * K + k0 + c8;
                    float4 x = *(const float4*)p;
                    float4 y = *(const float4*)(p + 4);
                    v0 = x.x; v1 = x.y; v2 = x.z; v3 = x.w;
                    v4 = y.x; v5 = y.y; v6 = y.z; v7 = y.w;
                    if (k0 + 32 < K)
                        __builtin_prefetch(p + 32, 0, 1);   // global_prefetch_b8
                } else {
                    v0 = v1 = v2 = v3 = v4 = v5 = v6 = v7 = 0.f;
                }
                uint4 pk;
                pk.x = packbf2(v0, v1); pk.y = packbf2(v2, v3);
                pk.z = packbf2(v4, v5); pk.w = packbf2(v6, v7);
                *(uint4*)(Alds + r * 40 + c8) = pk;
            } else {
                // ---- stage bf16 A tile: async DMA, 16B (8 bf16) per lane ----
                const int r   = t >> 2;           // 0..63
                const int seg = (t & 3) << 3;     // 0,8,16,24 (bf16 units)
                const int grow = rowBase + r;
                unsigned short* dst = Alds + r * 40 + seg;
                if (grow < M) {
                    const unsigned short* srcp = A2 + (long)grow * K + k0 + seg;
                    // Generic LDS pointer low 32 bits == LDS byte offset.
                    unsigned lds_off = (unsigned)(size_t)dst;
                    unsigned long long ga = (unsigned long long)(size_t)srcp;
                    asm volatile("global_load_async_to_lds_b128 %0, %1, off"
                                 :: "v"(lds_off), "v"(ga) : "memory");
                } else {
                    uint4 z = {0u, 0u, 0u, 0u};
                    *(uint4*)dst = z;
                }
            }
            // ---- stage W tile transposed: W[k0+k][colBase+n] -> Blds[n][k] ----
            {
                const int k  = t >> 3;            // 0..31
                const int n8 = (t & 7) << 3;      // 0..56
                const float* p = W + (long)(k0 + k) * N + colBase + n8;
                float4 x = *(const float4*)p;
                float4 y = *(const float4*)(p + 4);
                Blds[(n8 + 0) * 40 + k] = f2bf(x.x);
                Blds[(n8 + 1) * 40 + k] = f2bf(x.y);
                Blds[(n8 + 2) * 40 + k] = f2bf(x.z);
                Blds[(n8 + 3) * 40 + k] = f2bf(x.w);
                Blds[(n8 + 4) * 40 + k] = f2bf(y.x);
                Blds[(n8 + 5) * 40 + k] = f2bf(y.y);
                Blds[(n8 + 6) * 40 + k] = f2bf(y.z);
                Blds[(n8 + 7) * 40 + k] = f2bf(y.w);
            }
            if (phase == 1) {
                // Wait for this wave's async-LDS DMAs before the barrier.
                asm volatile("s_wait_asynccnt 0x0" ::: "memory");
            }
            __syncthreads();

            // ---- A fragment (contiguous k = 16*half + s, same order as B) ----
            Frag16 a;
            const unsigned short* ap = Alds + (waveR * 16 + l16) * 40 + half * 16;
            a.q[0] = *(const uint4*)ap;          // K 0..7   (half=0) / 16..23 (half=1)
            a.q[1] = *(const uint4*)(ap + 8);    // K 8..15  (half=0) / 24..31 (half=1)

            // ---- B fragments + WMMA ----
            {
                Frag16 b;
                const unsigned short* bp =
                    Blds + (waveC * 32 + l16) * 40 + half * 16;
                b.q[0] = *(const uint4*)bp;
                b.q[1] = *(const uint4*)(bp + 8);
                acc0 = __builtin_amdgcn_wmma_f32_16x16x32_bf16(
                    false, a.bf, false, b.bf, (short)0, acc0, false, false);
            }
            {
                Frag16 b;
                const unsigned short* bp =
                    Blds + (waveC * 32 + 16 + l16) * 40 + half * 16;
                b.q[0] = *(const uint4*)bp;
                b.q[1] = *(const uint4*)(bp + 8);
                acc1 = __builtin_amdgcn_wmma_f32_16x16x32_bf16(
                    false, a.bf, false, b.bf, (short)0, acc1, false, false);
            }
            __syncthreads();
        }
    }

    // ---- epilogue: bias + activation + bf16 store (C layout: M = 8*half+j) ----
    #pragma unroll
    for (int s2 = 0; s2 < 2; ++s2) {
        v8f acc = s2 ? acc1 : acc0;
        const int col = colBase + waveC * 32 + s2 * 16 + l16;
        const float bv = bias ? bias[col] : 0.f;
        #pragma unroll
        for (int j = 0; j < 8; ++j) {
            const int grow = rowBase + waveR * 16 + half * 8 + j;
            if (grow < M) {
                float x = acc[j] + bv;
                if (slope >= 0.f) x = (x > 0.f) ? x : slope * x;
                C[(long)grow * N + col] = f2bf(x);
            }
        }
    }
}

// ---------------------------------------------------------------------------
// Edge scatter: 64 lanes per edge, 8B bf16 gather -> fp32 atomic scatter.
// ---------------------------------------------------------------------------
__global__ void __launch_bounds__(256)
scatter_mean(const unsigned short* __restrict__ src, const int* __restrict__ es,
             const int* __restrict__ ed, float* __restrict__ agg,
             float* __restrict__ cnt, int E)
{
    const long tid = (long)blockIdx.x * blockDim.x + threadIdx.x;
    const int e = (int)(tid >> 6);
    const int l = (int)(tid & 63);
    if (e >= E) return;
    const int s = es[e];
    const int d = ed[e];
    uint2 v = *(const uint2*)(src + (long)s * 256 + l * 4);  // 4 bf16
    float* ap = agg + (long)d * 256 + l * 4;
    atomicAdd(ap + 0, bflo(v.x));
    atomicAdd(ap + 1, bfhi(v.x));
    atomicAdd(ap + 2, bflo(v.y));
    atomicAdd(ap + 3, bfhi(v.y));
    if (l == 0) atomicAdd(cnt + d, 1.0f);
}

__global__ void __launch_bounds__(256)
zero_f32(float* __restrict__ p, long n)
{
    long i = (long)blockIdx.x * blockDim.x + threadIdx.x;
    const long stride = (long)gridDim.x * blockDim.x;
    for (; i < n; i += stride) p[i] = 0.f;
}

// One block (256 threads) per row: agg[row] /= max(cnt[row], 1)
__global__ void __launch_bounds__(256)
norm_rows(float* __restrict__ agg, const float* __restrict__ cnt, int rows)
{
    const int row = blockIdx.x;
    if (row >= rows) return;
    const float inv = 1.0f / fmaxf(cnt[row], 1.0f);
    agg[(long)row * 256 + threadIdx.x] *= inv;
}

// out[m][0..1] = f_bf16[m] . linW[:,0..1] + b ; one wave (32 lanes) per row.
__global__ void __launch_bounds__(256)
final_linear(const unsigned short* __restrict__ f, const float* __restrict__ W,
             const float* __restrict__ b, float* __restrict__ out, int M)
{
    const int row  = blockIdx.x * 8 + (threadIdx.x >> 5);
    const int lane = threadIdx.x & 31;
    if (row >= M) return;
    const uint4* fp = (const uint4*)(f + (long)row * 256);
    uint4 q = fp[lane];                      // 8 bf16, k = lane*8 .. lane*8+7
    const int k = lane * 8;
    float s0, s1;
    s0  = bflo(q.x) * W[(k + 0) * 2 + 0] + bfhi(q.x) * W[(k + 1) * 2 + 0];
    s1  = bflo(q.x) * W[(k + 0) * 2 + 1] + bfhi(q.x) * W[(k + 1) * 2 + 1];
    s0 += bflo(q.y) * W[(k + 2) * 2 + 0] + bfhi(q.y) * W[(k + 3) * 2 + 0];
    s1 += bflo(q.y) * W[(k + 2) * 2 + 1] + bfhi(q.y) * W[(k + 3) * 2 + 1];
    s0 += bflo(q.z) * W[(k + 4) * 2 + 0] + bfhi(q.z) * W[(k + 5) * 2 + 0];
    s1 += bflo(q.z) * W[(k + 4) * 2 + 1] + bfhi(q.z) * W[(k + 5) * 2 + 1];
    s0 += bflo(q.w) * W[(k + 6) * 2 + 0] + bfhi(q.w) * W[(k + 7) * 2 + 0];
    s1 += bflo(q.w) * W[(k + 6) * 2 + 1] + bfhi(q.w) * W[(k + 7) * 2 + 1];
    #pragma unroll
    for (int off = 16; off > 0; off >>= 1) {
        s0 += __shfl_xor(s0, off, 32);
        s1 += __shfl_xor(s1, off, 32);
    }
    if (lane == 0) {
        out[(long)row * 2 + 0] = s0 + b[0];
        out[(long)row * 2 + 1] = s1 + b[1];
    }
}

// ---------------------------------------------------------------------------
extern "C" void kernel_launch(void* const* d_in, const int* in_sizes, int n_in,
                              void* d_out, int out_size, void* d_ws, size_t ws_size,
                              hipStream_t stream)
{
    const float* x_host = (const float*)d_in[0];
    const float* x_flow = (const float*)d_in[1];
    const int*   hf_src = (const int*)d_in[2];
    const int*   hf_dst = (const int*)d_in[3];
    const int*   fh_src = (const int*)d_in[4];
    const int*   fh_dst = (const int*)d_in[5];
    const float* hpW    = (const float*)d_in[6];
    const float* hpb    = (const float*)d_in[7];
    const float* fpW    = (const float*)d_in[8];
    const float* fpb    = (const float*)d_in[9];
    const float* Wl_hf  = (const float*)d_in[10];
    const float* bl_hf  = (const float*)d_in[11];
    const float* Wr_hf  = (const float*)d_in[12];
    const float* Wl_fh  = (const float*)d_in[13];
    const float* bl_fh  = (const float*)d_in[14];
    const float* Wr_fh  = (const float*)d_in[15];
    const float* linW   = (const float*)d_in[16];
    const float* linb   = (const float*)d_in[17];
    float* out = (float*)d_out;

    const int DHOST = 64, DFLOW = 128, DH = 256, NLAYERS = 2;
    const int n_host = in_sizes[0] / DHOST;
    const int n_flow = in_sizes[1] / DFLOW;
    const int E      = in_sizes[2];

    // Workspace: fp32 agg + cnt first, then bf16 activation ping-pong buffers.
    float* agg = (float*)d_ws;
    float* cnt = agg + (size_t)n_flow * DH;
    unsigned short* h0 = (unsigned short*)(cnt + n_flow);
    unsigned short* h1 = h0 + (size_t)n_host * DH;
    unsigned short* f0 = h1 + (size_t)n_host * DH;
    unsigned short* f1 = f0 + (size_t)n_flow * DH;

    const dim3 blk(256);
    const unsigned scatter_grid = (unsigned)(((long)E * 64 + 255) / 256);

    // Projections: h = relu(x_host @ hpW + hpb), f = relu(x_flow @ fpW + fpb)
    wmma_gemm_bias_act<<<dim3(DH / 64, (n_host + 63) / 64), blk, 0, stream>>>(
        x_host, hpW, DHOST, nullptr, nullptr, 0, hpb, h0, n_host, DH, 0.0f);
    wmma_gemm_bias_act<<<dim3(DH / 64, (n_flow + 63) / 64), blk, 0, stream>>>(
        x_flow, fpW, DFLOW, nullptr, nullptr, 0, fpb, f0, n_flow, DH, 0.0f);

    unsigned short* hin = h0; unsigned short* hout = h1;
    unsigned short* fin = f0; unsigned short* fout = f1;

    for (int l = 0; l < NLAYERS; ++l) {
        const float* wl_hf = Wl_hf + (size_t)l * DH * DH;
        const float* wr_hf = Wr_hf + (size_t)l * DH * DH;
        const float* wl_fh = Wl_fh + (size_t)l * DH * DH;
        const float* wr_fh = Wr_fh + (size_t)l * DH * DH;
        const float* b_hf  = bl_hf + (size_t)l * DH;
        const float* b_fh  = bl_fh + (size_t)l * DH;

        // host -> flow: agg_f = mean(h over hf edges); f' = leaky(agg@Wl + f@Wr + b)
        zero_f32<<<4096, blk, 0, stream>>>(agg, (long)n_flow * DH);
        zero_f32<<<256, blk, 0, stream>>>(cnt, (long)n_flow);
        scatter_mean<<<scatter_grid, blk, 0, stream>>>(hin, hf_src, hf_dst, agg, cnt, E);
        norm_rows<<<n_flow, blk, 0, stream>>>(agg, cnt, n_flow);
        wmma_gemm_bias_act<<<dim3(DH / 64, (n_flow + 63) / 64), blk, 0, stream>>>(
            agg, wl_hf, DH, fin, wr_hf, DH, b_hf, fout, n_flow, DH, 0.01f);

        // flow -> host: agg_h = mean(f over fh edges); h' = leaky(agg@Wl + h@Wr + b)
        zero_f32<<<4096, blk, 0, stream>>>(agg, (long)n_host * DH);
        zero_f32<<<256, blk, 0, stream>>>(cnt, (long)n_host);
        scatter_mean<<<scatter_grid, blk, 0, stream>>>(fin, fh_src, fh_dst, agg, cnt, E);
        norm_rows<<<n_host, blk, 0, stream>>>(agg, cnt, n_host);
        wmma_gemm_bias_act<<<dim3(DH / 64, (n_host + 63) / 64), blk, 0, stream>>>(
            agg, wl_fh, DH, hin, wr_fh, DH, b_fh, hout, n_host, DH, 0.01f);

        unsigned short* tmp;
        tmp = hin; hin = hout; hout = tmp;
        tmp = fin; fin = fout; fout = tmp;
    }

    final_linear<<<(n_flow + 7) / 8, blk, 0, stream>>>(fin, linW, linb, out, n_flow);
}
